// Net_59021440581865
// MI455X (gfx1250) — matrix operand order repaired and checked by reading, not compiled, not run on hardware
//
#include <hip/hip_runtime.h>
#include <hip/hip_bf16.h>
#include <string.h>

// ---------------------------------------------------------------------------
// Fused 3-layer LSTM (H=64) + DeepAR Gaussian heads for MI455X (gfx1250).
// One persistent workgroup per 16-row batch tile; everything stays on-chip.
// Recurrent matmuls use V_WMMA_F32_16X16X32_BF16 (bf16 in, f32 accumulate).
// ---------------------------------------------------------------------------

#define Tn   1024
#define Nn   512
#define Hd   64
#define DX   8

typedef __attribute__((ext_vector_type(16))) __bf16 v16bf;
typedef __attribute__((ext_vector_type(8)))  float  v8f;

union BFrag {                 // one 16x32 (A) or 32x16 (B) bf16 WMMA fragment
    v16bf        v;           // for the wmma builtin
    unsigned int u[8];        // for packing
    uint4        q[2];        // for 128-bit LDS moves
};

// ---- LDS layout (dwords) ---------------------------------------------------
// B-fragments of [W_ih^T ; W_hh^T] per layer: tiles indexed (ct, kt),
// 32 lanes x 8 dwords per tile.
#define L0_NKT 3              // layer0 K = 32 (x padded 8->32) + 64 (h)
#define L1_NKT 4              // layers 1/2: K = 64 + 64
#define L0_DW  (16 * L0_NKT * 32 * 8)   // 12288
#define L1_DW  (16 * L1_NKT * 32 * 8)   // 16384
#define W0_OFF 0
#define W1_OFF (W0_OFF + L0_DW)         // 12288
#define W2_OFF (W1_OFF + L1_DW)         // 28672
#define H0_OFF (W2_OFF + L1_DW)         // 45056  (16x64 f32 h buffer)
#define H1_OFF (H0_OFF + 1024)          // 46080
#define H2_OFF (H1_OFF + 1024)          // 47104
#define WM_OFF (H2_OFF + 1024)          // 48128
#define WA_OFF (WM_OFF + 64)            // 48192
#define SMEM_DW (WA_OFF + 64)           // 48256 dwords = 193024 bytes

#define L0_RUNS (16 * L0_NKT * 32 * 2)  // 3072 8-element packing runs
#define L1_RUNS (16 * L1_NKT * 32 * 2)  // 4096
#define TOT_RUNS (L0_RUNS + 2 * L1_RUNS)

// ---- helpers ---------------------------------------------------------------
// Pack two f32 -> packed bf16 dword with ONE v_perm_b32 (truncating convert:
// take the top 16 bits of each float).  sel 0x07060302: D[15:0]=lo[31:16],
// D[31:16]=hi[31:16]  (V_PERM_B32: S1 supplies pool bytes 0-3, S0 bytes 4-7).
static __device__ __forceinline__ unsigned pk(float lo, float hi) {
    return __builtin_amdgcn_perm(__builtin_bit_cast(unsigned, hi),
                                 __builtin_bit_cast(unsigned, lo),
                                 0x07060302u);
}
static __device__ __forceinline__ float frcp(float x) {   // raw v_rcp_f32
    return __builtin_amdgcn_rcpf(x);
}
static __device__ __forceinline__ float fsigmoid(float x) {
    return frcp(1.0f + __expf(-x));                       // 1 trans exp + 1 trans rcp
}
static __device__ __forceinline__ float ftanh(float x) {
    // saturates cleanly: exp->inf gives 1, exp->0 gives -1
    return 1.0f - 2.0f * frcp(__expf(2.0f * x) + 1.0f);
}
static __device__ __forceinline__ float softplusf(float x) {
    float sp = __logf(1.0f + __expf(-fabsf(x)));
    return (x > 0.0f) ? (x + sp) : sp;
}

// Build the two K-tile A-fragments of a 16x64 h tile held in LDS (f32).
// A 16x32 bf16 layout: lane L -> row M=L&15, kh=L>>4;
// VGPR0-3 hold K = kh*8 + {0..7}, VGPR4-7 hold K = 16 + kh*8 + {0..7}.
static __device__ __forceinline__ void build_afrag(const float* __restrict__ hbuf,
                                                   BFrag* fr, int lane) {
    const int rowA = lane & 15, kh = lane >> 4;
    const float* hp = hbuf + rowA * Hd;
#pragma unroll
    for (int kt = 0; kt < 2; ++kt) {
        const int kA = kt * 32 + kh * 8;
        float4 a0 = *(const float4*)(hp + kA);
        float4 a1 = *(const float4*)(hp + kA + 4);
        float4 b0 = *(const float4*)(hp + kA + 16);
        float4 b1 = *(const float4*)(hp + kA + 20);
        fr[kt].u[0] = pk(a0.x, a0.y); fr[kt].u[1] = pk(a0.z, a0.w);
        fr[kt].u[2] = pk(a1.x, a1.y); fr[kt].u[3] = pk(a1.z, a1.w);
        fr[kt].u[4] = pk(b0.x, b0.y); fr[kt].u[5] = pk(b0.z, b0.w);
        fr[kt].u[6] = pk(b1.x, b1.y); fr[kt].u[7] = pk(b1.z, b1.w);
    }
}

// Store this wave's 16x16 h sub-tile (C/D layout: lane=col, VGPR=row) to LDS f32.
static __device__ __forceinline__ void store_h_lds(float* __restrict__ hbuf,
                                                   const float* hout, int wave, int lane) {
    const int col = wave * 16 + (lane & 15);
    const int m0  = (lane >> 4) * 8;
#pragma unroll
    for (int i = 0; i < 8; ++i) hbuf[(m0 + i) * Hd + col] = hout[i];
}

// One LSTM layer timestep for this wave: 4 gate tiles x NKT WMMA, then i/f/g/o.
// Gate bias is folded into the WMMA accumulator init (lane = gate column).
template <int NKT>
static __device__ __forceinline__ void layer_step(const unsigned* __restrict__ smem,
                                                  int wbase, const BFrag* afr,
                                                  const float* bias, float* cst,
                                                  float* hout, int wave, int lane) {
    v8f acc[4];
#pragma unroll
    for (int g = 0; g < 4; ++g) {
        v8f a;
#pragma unroll
        for (int i = 0; i < 8; ++i) a[i] = bias[g];   // C := bias (per-column)
        const int ct = 4 * g + wave;                  // gate tile: cols 64g + 16*wave
#pragma unroll
        for (int kt = 0; kt < NKT; ++kt) {
            BFrag bf;
            const uint4* p = (const uint4*)&smem[wbase + ((ct * NKT + kt) * 32 + lane) * 8];
            bf.q[0] = p[0]; bf.q[1] = p[1];           // ds_load_b128 x2
            a = __builtin_amdgcn_wmma_f32_16x16x32_bf16(
                    false, afr[kt].v, false, bf.v, (short)0, a, false, false);
        }
        acc[g] = a;
    }
#pragma unroll
    for (int i = 0; i < 8; ++i) {
        float ig = fsigmoid(acc[0][i]);
        float fg = fsigmoid(acc[1][i]);
        float gg = ftanh   (acc[2][i]);
        float og = fsigmoid(acc[3][i]);
        float c  = fg * cst[i] + ig * gg;
        cst[i]   = c;
        hout[i]  = og * ftanh(c);
    }
}

__global__ __launch_bounds__(128) void lstm3_deepar_fused(
    const float* __restrict__ x,   const float* __restrict__ v,
    const float* __restrict__ wih0, const float* __restrict__ whh0, const float* __restrict__ b0,
    const float* __restrict__ wih1, const float* __restrict__ whh1, const float* __restrict__ b1,
    const float* __restrict__ wih2, const float* __restrict__ whh2, const float* __restrict__ b2,
    const float* __restrict__ wm, const float* __restrict__ bm,
    const float* __restrict__ wa, const float* __restrict__ ba,
    float* __restrict__ out) {
    extern __shared__ unsigned smem[];
    const int tid  = threadIdx.x;
    const int lane = tid & 31;
    const int wave = tid >> 5;          // hidden-col group 0..3
    const int row0 = blockIdx.x * 16;   // batch-row tile

    // ---- one-time: pack [W_ih^T ; W_hh^T] into bf16 B-fragments in LDS -----
    // B 32x16 bf16 layout: lane L -> col n=L&15, kh=L>>4; dword i = K pair
    // (kh*8 + 2i, +1) for i<4, (16 + kh*8 + 2(i-4), +1) for i>=4, within the 32-wide K tile.
    for (int r = tid; r < TOT_RUNS; r += 128) {
        int rl, nkt, Dreal, Dpad, base;
        const float *WI, *WH;
        if (r < L0_RUNS) {
            rl = r; WI = wih0; WH = whh0; Dreal = DX; Dpad = 32; nkt = L0_NKT; base = W0_OFF;
        } else if (r < L0_RUNS + L1_RUNS) {
            rl = r - L0_RUNS; WI = wih1; WH = whh1; Dreal = Hd; Dpad = Hd; nkt = L1_NKT; base = W1_OFF;
        } else {
            rl = r - L0_RUNS - L1_RUNS; WI = wih2; WH = whh2; Dreal = Hd; Dpad = Hd; nkt = L1_NKT; base = W2_OFF;
        }
        const int tile = rl >> 6;           // (ct * nkt + kt)
        const int rem  = rl & 63;
        const int ln   = rem >> 1;          // lane within tile
        const int half = rem & 1;           // dwords 0-3 vs 4-7
        const int ct   = tile / nkt;
        const int kt   = tile - ct * nkt;
        const int n    = ct * 16 + (ln & 15);
        const int ks   = kt * 32 + (ln >> 4) * 8 + half * 16;
        float f[8];
#pragma unroll
        for (int j = 0; j < 8; ++j) {
            const int k = ks + j;
            f[j] = (k < Dpad) ? ((k < Dreal) ? WI[n * Dreal + k] : 0.0f)
                              : WH[n * Hd + (k - Dpad)];
        }
        uint4* dst = (uint4*)&smem[base + (tile * 32 + ln) * 8 + half * 4];
        *dst = make_uint4(pk(f[0], f[1]), pk(f[2], f[3]), pk(f[4], f[5]), pk(f[6], f[7]));
    }
    if (tid < 64)       smem[WM_OFF + tid]        = __builtin_bit_cast(unsigned, wm[tid]);
    else if (tid < 128) smem[WA_OFF + (tid - 64)] = __builtin_bit_cast(unsigned, wa[tid - 64]);

    // per-lane gate biases: this lane's gate column is 16*wave + (lane&15)
    const int cc = wave * 16 + (lane & 15);
    float bs0[4], bs1[4], bs2[4];
#pragma unroll
    for (int g = 0; g < 4; ++g) {
        bs0[g] = b0[g * Hd + cc];
        bs1[g] = b1[g * Hd + cc];
        bs2[g] = b2[g * Hd + cc];
    }
    // head setup (wave0 -> mean, wave1 -> softplus std), 16 rows on lanes 0-15
    float vrow = 0.0f, hb = 0.0f;
    if (wave < 2 && lane < 16) {
        vrow = v[row0 + lane];
        hb   = (wave == 0) ? bm[0] : ba[0];
    }

    float* h0b = (float*)&smem[H0_OFF];
    float* h1b = (float*)&smem[H1_OFF];
    float* h2b = (float*)&smem[H2_OFF];

    __syncthreads();

    // recurrent state
    BFrag hf0[2], hf1[2], hf2[2];
#pragma unroll
    for (int k = 0; k < 2; ++k)
#pragma unroll
        for (int j = 0; j < 8; ++j) { hf0[k].u[j] = 0; hf1[k].u[j] = 0; hf2[k].u[j] = 0; }
    float c0[8] = {}, c1[8] = {}, c2[8] = {};
    float h0o[8], h1o[8], h2o[8];

    const float* xrow = x + ((size_t)(row0 + (lane & 15)) * Tn) * DX;

    for (int t = 0; t < Tn; ++t) {
        // ---- x_t A-fragment (K padded 8->32; lanes 16-31 hold only pad) ----
        BFrag ax;
#pragma unroll
        for (int j = 0; j < 8; ++j) ax.u[j] = 0;
        if (lane < 16) {
            const float4* xp = (const float4*)(xrow + (size_t)t * DX);
            float4 a = xp[0], b = xp[1];                 // global_load_b128 x2
            ax.u[0] = pk(a.x, a.y); ax.u[1] = pk(a.z, a.w);
            ax.u[2] = pk(b.x, b.y); ax.u[3] = pk(b.z, b.w);
            if (t + 8 < Tn)
                __builtin_prefetch(xrow + (size_t)(t + 8) * DX, 0, 1); // global_prefetch_b8
        }

        // ---- layer 0: gates = [x | h0] @ W0, 12 WMMA / wave ----
        BFrag a0[3] = {ax, hf0[0], hf0[1]};
        layer_step<L0_NKT>(smem, W0_OFF, a0, bs0, c0, h0o, wave, lane);
        store_h_lds(h0b, h0o, wave, lane);
        __syncthreads();
        build_afrag(h0b, hf0, lane);

        // ---- layer 1: gates = [h0 | h1] @ W1, 16 WMMA / wave ----
        BFrag a1[4] = {hf0[0], hf0[1], hf1[0], hf1[1]};
        layer_step<L1_NKT>(smem, W1_OFF, a1, bs1, c1, h1o, wave, lane);
        store_h_lds(h1b, h1o, wave, lane);
        __syncthreads();
        build_afrag(h1b, hf1, lane);

        // ---- layer 2: gates = [h1 | h2] @ W2, 16 WMMA / wave ----
        BFrag a2[4] = {hf1[0], hf1[1], hf2[0], hf2[1]};
        layer_step<L1_NKT>(smem, W2_OFF, a2, bs2, c2, h2o, wave, lane);
        store_h_lds(h2b, h2o, wave, lane);
        __syncthreads();
        build_afrag(h2b, hf2, lane);

        // ---- heads: m = (h2.wm + bm)*v ; a = softplus(h2.wa + ba)*v --------
        if (wave < 2 && lane < 16) {
            const float* hr = h2b + lane * Hd;
            const float* wv = (const float*)&smem[(wave == 0) ? WM_OFF : WA_OFF];
            float s = hb;
#pragma unroll
            for (int k = 0; k < Hd; k += 4) {
                float4 hq = *(const float4*)(hr + k);
                float4 wq = *(const float4*)(wv + k);
                s += hq.x * wq.x + hq.y * wq.y + hq.z * wq.z + hq.w * wq.w;
            }
            float r = (wave == 0) ? (s * vrow) : (softplusf(s) * vrow);
            out[(size_t)wave * Nn * Tn + (size_t)(row0 + lane) * Tn + t] = r;
        }
        // next iteration's first h-buffer write is ordered after the next
        // __syncthreads(); the sync after layer0's store protects h2b readers.
    }
}

extern "C" void kernel_launch(void* const* d_in, const int* in_sizes, int n_in,
                              void* d_out, int out_size, void* d_ws, size_t ws_size,
                              hipStream_t stream) {
    const float* x    = (const float*)d_in[0];
    const float* v    = (const float*)d_in[1];
    const float* wih0 = (const float*)d_in[2];
    const float* whh0 = (const float*)d_in[3];
    const float* b0   = (const float*)d_in[4];
    const float* wih1 = (const float*)d_in[5];
    const float* whh1 = (const float*)d_in[6];
    const float* b1   = (const float*)d_in[7];
    const float* wih2 = (const float*)d_in[8];
    const float* whh2 = (const float*)d_in[9];
    const float* b2   = (const float*)d_in[10];
    const float* wm   = (const float*)d_in[11];
    const float* bm   = (const float*)d_in[12];
    const float* wa   = (const float*)d_in[13];
    const float* ba   = (const float*)d_in[14];
    float* out = (float*)d_out;

    const size_t smem_bytes = (size_t)SMEM_DW * 4;  // 193,024 B (<320 KB/WGP)
    (void)hipFuncSetAttribute((const void*)lstm3_deepar_fused,
                              hipFuncAttributeMaxDynamicSharedMemorySize,
                              (int)smem_bytes);
    lstm3_deepar_fused<<<dim3(Nn / 16), dim3(128), smem_bytes, stream>>>(
        x, v, wih0, whh0, b0, wih1, whh1, b1, wih2, whh2, b2, wm, bm, wa, ba, out);
}
